// SelfAtten_17686675325048
// MI455X (gfx1250) — matrix operand order repaired and checked by reading, not compiled, not run on hardware
//
#include <hip/hip_runtime.h>

#define DEV __device__ __forceinline__

typedef __attribute__((ext_vector_type(16))) __bf16 bf16x16;
typedef __attribute__((ext_vector_type(8)))  float  f32x8;
typedef __attribute__((address_space(3))) unsigned short lds_us;

union FragAB { unsigned int u[8]; uint4 q[2]; bf16x16 v; };
union FragC  { float f[8];        f32x8  v; };

DEV unsigned int pack2_bf16(float lo, float hi) {
  unsigned int a = __builtin_bit_cast(unsigned int, lo);
  unsigned int b = __builtin_bit_cast(unsigned int, hi);
  a = (a + 0x7FFFu + ((a >> 16) & 1u)) >> 16;
  b = (b + 0x7FFFu + ((b >> 16) & 1u)) >> 16;
  return (a & 0xFFFFu) | (b << 16);
}

DEV unsigned short f2bf(float f) {
  unsigned int x = __builtin_bit_cast(unsigned int, f);
  x = (x + 0x7FFFu + ((x >> 16) & 1u)) >> 16;
  return (unsigned short)x;
}

// A-frag (16x32 bf16, ISA 7.12.2): two contiguous 16B groups per lane.
// rp = pointer to element 0 of this lane's row (k-chunk base already added).
DEV void load_a_frag(FragAB& a, const unsigned short* rp, int lh) {
  a.q[0] = *(const uint4*)(rp + 8 * lh);
  a.q[1] = *(const uint4*)(rp + 16 + 8 * lh);
}
// B-frag (32x16 bf16): two contiguous 16B groups per lane (same k order as A).
DEV void load_b_frag(FragAB& b, const unsigned short* rp, int lh) {
  b.q[0] = *(const uint4*)(rp + 8 * lh);
  b.q[1] = *(const uint4*)(rp + 16 + 8 * lh);
}

// ---------------------------------------------------------------------------
// Projection: out = X @ W^T + bias, emitted as bf16.
// TRANSPOSED==0: out [16384, 512] row-major (Q, also used as K)
// TRANSPOSED==1: out [4, 512, 4096] (V^T per batch; packed uint4 stores)
// ---------------------------------------------------------------------------
template <int TRANSPOSED>
__global__ __launch_bounds__(256) void proj_kernel(
    const float* __restrict__ X,    // [16384, 512]
    const float* __restrict__ W,    // [512, 512]
    const float* __restrict__ bias, // [512]
    unsigned short* __restrict__ out) {
  constexpr int MB = 128, KC = 32;
  constexpr int LS = 40; // LDS row stride (bf16) -> 80B rows, 16B aligned
  __shared__ alignas(16) unsigned short Xl[MB * LS];
  __shared__ alignas(16) unsigned short Wl[64 * LS];

  const int tid  = threadIdx.x;
  const int wave = tid >> 5;
  const int lane = tid & 31;
  const int ln   = lane & 15;
  const int lh   = lane >> 4;

  const int m0 = blockIdx.x * MB; // gridDim.x = 128
  const int e0 = blockIdx.y * 64; // gridDim.y = 8

  FragC acc[4];
#pragma unroll
  for (int t = 0; t < 4; ++t)
#pragma unroll
    for (int i = 0; i < 8; ++i) acc[t].f[i] = 0.f;

  for (int k0 = 0; k0 < 512; k0 += KC) {
    __syncthreads();
#pragma unroll
    for (int i = 0; i < 4; ++i) {           // X tile 128x32 f32 -> bf16
      int j = tid + i * 256;                // float4 id 0..1023
      int r = j >> 3;
      int c = (j & 7) * 4;
      float4 xv = *(const float4*)(X + (size_t)(m0 + r) * 512 + k0 + c);
      *(unsigned int*)(&Xl[r * LS + c])     = pack2_bf16(xv.x, xv.y);
      *(unsigned int*)(&Xl[r * LS + c + 2]) = pack2_bf16(xv.z, xv.w);
    }
#pragma unroll
    for (int i = 0; i < 2; ++i) {           // W tile 64x32 f32 -> bf16
      int j = tid + i * 256;
      int r = j >> 3;
      int c = (j & 7) * 4;
      float4 wv = *(const float4*)(W + (size_t)(e0 + r) * 512 + k0 + c);
      *(unsigned int*)(&Wl[r * LS + c])     = pack2_bf16(wv.x, wv.y);
      *(unsigned int*)(&Wl[r * LS + c + 2]) = pack2_bf16(wv.z, wv.w);
    }
    __syncthreads();

    FragAB a;
    load_a_frag(a, &Xl[(wave * 16 + ln) * LS], lh);
#pragma unroll
    for (int t = 0; t < 4; ++t) {
      FragAB bf;
      load_b_frag(bf, &Wl[(t * 16 + ln) * LS], lh);
      acc[t].v = __builtin_amdgcn_wmma_f32_16x16x32_bf16(
          false, a.v, false, bf.v, (short)0, acc[t].v, false, false);
    }
  }

#pragma unroll
  for (int t = 0; t < 4; ++t) {
    int e = e0 + t * 16 + ln;
    float bv = bias[e];
    if (TRANSPOSED) {
      // lane's 8 outputs hit 8 consecutive tokens of V^T row e -> one uint4
      int m = m0 + wave * 16 + 8 * lh;      // token base (mult of 8)
      int b = m >> 12, tok = m & 4095;
      uint4 pk;
      pk.x = pack2_bf16(acc[t].f[0] + bv, acc[t].f[1] + bv);
      pk.y = pack2_bf16(acc[t].f[2] + bv, acc[t].f[3] + bv);
      pk.z = pack2_bf16(acc[t].f[4] + bv, acc[t].f[5] + bv);
      pk.w = pack2_bf16(acc[t].f[6] + bv, acc[t].f[7] + bv);
      *(uint4*)(out + ((size_t)b * 512 + e) * 4096 + tok) = pk;
    } else {
#pragma unroll
      for (int v = 0; v < 8; ++v) {
        int m = m0 + wave * 16 + v + 8 * lh;
        out[(size_t)m * 512 + e] = f2bf(acc[t].f[v] + bv);
      }
    }
  }
}

// ---------------------------------------------------------------------------
// Fused sigmoid-attention: Out[b, q, :] = sum_keys sigmoid(q.k/sqrt(D)) * v
// One block = 16 queries, 8 waves. Wave w computes scores for its own 32-key
// chunk -> LDS P; then all waves accumulate their 64 output columns.
// ---------------------------------------------------------------------------
__global__ __launch_bounds__(256) void attn_kernel(
    const unsigned short* __restrict__ Q,   // bf16 [16384, 512] (also K)
    const unsigned short* __restrict__ Vt,  // bf16 [4, 512, 4096]
    float* __restrict__ Out) {              // f32  [4, 4096, 512]
  constexpr int QS = 520;  // 1040B rows (16B aligned)
  constexpr int PS = 40;   // 80B rows (16B aligned, conflict-free frag reads)
  __shared__ alignas(16) unsigned short Ql[16 * QS];
  __shared__ alignas(16) unsigned short Pl[8][16 * PS];

  const int tid  = threadIdx.x;
  const int wave = tid >> 5;
  const int lane = tid & 31;
  const int ln   = lane & 15;
  const int lh   = lane >> 4;

  const int tileid = blockIdx.x;       // 0..1023
  const int b      = tileid >> 8;
  const int q0     = (tileid & 255) * 16;

  const unsigned short* Qb = Q  + (size_t)b * 4096 * 512;
  const unsigned short* Vb = Vt + (size_t)b * 512 * 4096;

  // stage Q tile 16x512 bf16 via async global->LDS (ASYNCcnt-tracked)
#pragma unroll
  for (int i = 0; i < 4; ++i) {
    int j = tid + i * 256;   // 16B packet id 0..1023 (64 per row)
    int r = j >> 6;
    int c = (j & 63) * 8;
    const unsigned short* g = Qb + (size_t)(q0 + r) * 512 + c;
    unsigned int lds_off =
        (unsigned int)(unsigned long long)(lds_us*)(&Ql[r * QS + c]);
    asm volatile("global_load_async_to_lds_b128 %0, %1, off"
                 :: "v"(lds_off), "v"(g) : "memory");
  }
  asm volatile("s_wait_asynccnt 0x0" ::: "memory");

  FragC oacc[4];
#pragma unroll
  for (int t = 0; t < 4; ++t)
#pragma unroll
    for (int i = 0; i < 8; ++i) oacc[t].f[i] = 0.f;

  const int col0 = wave * 64;
  const float scale = 0.0441941738f; // 1/sqrt(512)

  __syncthreads();

  for (int it = 0; it < 16; ++it) {
    // ---- S phase: this wave's 32-key chunk ----
    const int kb = (it * 8 + wave) * 32;
    FragC s[2];
#pragma unroll
    for (int t = 0; t < 2; ++t)
#pragma unroll
      for (int i = 0; i < 8; ++i) s[t].f[i] = 0.f;

    for (int k0 = 0; k0 < 512; k0 += 32) {
      FragAB a;
      load_a_frag(a, &Ql[ln * QS + k0], lh);
#pragma unroll
      for (int t = 0; t < 2; ++t) {
        FragAB bf;  // K rows (keys): two 16B groups per lane, L2-resident
        load_b_frag(bf, Qb + (size_t)(kb + t * 16 + ln) * 512 + k0, lh);
        s[t].v = __builtin_amdgcn_wmma_f32_16x16x32_bf16(
            false, a.v, false, bf.v, (short)0, s[t].v, false, false);
      }
    }

    // sigmoid -> bf16 P tile in LDS (C layout: row = v+8*lh, col = t*16+ln)
#pragma unroll
    for (int t = 0; t < 2; ++t)
#pragma unroll
      for (int v = 0; v < 8; ++v) {
        float p = 1.0f / (1.0f + __expf(-s[t].f[v] * scale));
        Pl[wave][(v + 8 * lh) * PS + t * 16 + ln] = f2bf(p);
      }
    __syncthreads();

    // ---- PV phase: this wave's 64 columns over all 8 chunks ----
#pragma unroll
    for (int c = 0; c < 8; ++c) {
      const int kb2 = (it * 8 + c) * 32;
      FragAB a;
      load_a_frag(a, &Pl[c][ln * PS], lh);
#pragma unroll
      for (int t = 0; t < 4; ++t) {
        FragAB bf;  // V^T rows = output cols: contiguous keys -> 2x uint4
        load_b_frag(bf, Vb + (size_t)(col0 + t * 16 + ln) * 4096 + kb2, lh);
        oacc[t].v = __builtin_amdgcn_wmma_f32_16x16x32_bf16(
            false, a.v, false, bf.v, (short)0, oacc[t].v, false, false);
      }
    }
    __syncthreads();
  }

  float* Ob = Out + (size_t)b * 4096 * 512;
#pragma unroll
  for (int t = 0; t < 4; ++t)
#pragma unroll
    for (int v = 0; v < 8; ++v)
      Ob[(size_t)(q0 + v + 8 * lh) * 512 + col0 + t * 16 + ln] = oacc[t].f[v];
}

extern "C" void kernel_launch(void* const* d_in, const int* in_sizes, int n_in,
                              void* d_out, int out_size, void* d_ws, size_t ws_size,
                              hipStream_t stream) {
  (void)in_sizes; (void)n_in; (void)out_size; (void)ws_size;
  const float* x  = (const float*)d_in[0];
  const float* Wq = (const float*)d_in[1];
  const float* bq = (const float*)d_in[2];
  const float* Wv = (const float*)d_in[3];
  const float* bv = (const float*)d_in[4];

  unsigned short* Qws  = (unsigned short*)d_ws;             // 16 MB bf16 Q(=K)
  unsigned short* Vtws = Qws + (size_t)16384 * 512;         // 16 MB bf16 V^T

  dim3 pg(128, 8), pb(256);
  proj_kernel<0><<<pg, pb, 0, stream>>>(x, Wq, bq, Qws);
  proj_kernel<1><<<pg, pb, 0, stream>>>(x, Wv, bv, Vtws);
  attn_kernel<<<dim3(1024), dim3(256), 0, stream>>>(Qws, Vtws, (float*)d_out);
}
